// TorchSpikingConv2dLayer_40123584479820
// MI455X (gfx1250) — matrix operand order repaired and checked by reading, not compiled, not run on hardware
//
#include <hip/hip_runtime.h>
#include <hip/hip_bf16.h>
#include <math.h>

typedef __attribute__((ext_vector_type(16))) _Float16 v16h;
typedef __attribute__((ext_vector_type(8)))  _Float16 v8h;
typedef __attribute__((ext_vector_type(2)))  _Float16 h2;
typedef __attribute__((ext_vector_type(8)))  float    v8f;

#define T_DIM  128
#define CIN    64
#define COUT   128
#define HW     64
#define CHW    (COUT * HW * HW)     // 524288, d_out stride per timestep
#define KPAD   16                   // 3x3 = 9 taps padded to 16 -> 2 ci per 32-wide K step
#define KSTEPS (CIN / 2)            // 32 WMMA K-steps per conv
#define QTOT   (T_DIM * KSTEPS)     // 4096 flattened (t, s) steps
#define SMSTR  40                   // LDS row stride (halves): 80B rows, 16B aligned, no conflicts

// ---------------------------------------------------------------------------
// Kernel 1: pack weights [co][ci][3][3] f32 -> [co][ci][16] f16 (taps 9..15 = 0).
// 256 KB total, L2-resident. Each WMMA B-fragment lane = one contiguous 32B read.
// ---------------------------------------------------------------------------
__global__ void pack_weights(const float* __restrict__ w, _Float16* __restrict__ wp) {
    int i = blockIdx.x * blockDim.x + threadIdx.x;       // COUT*CIN*KPAD = 131072
    if (i >= COUT * CIN * KPAD) return;
    int r  = i & 15;
    int ci = (i >> 4) & (CIN - 1);
    int co = i >> 10;
    float v = 0.0f;
    if (r < 9) v = w[((co * CIN + ci) * 3 + (r / 3)) * 3 + (r % 3)];
    wp[i] = (_Float16)v;
}

// ---------------------------------------------------------------------------
// Kernel 2: fused implicit-GEMM conv + integrate/subtract-and-fire scan.
// Block: 128 thr = 4 waves, owns 16 pixels (fixed h, w0) x 64 co (co-half),
// loops t = 0..127 serially; membrane state lives in registers next to the
// WMMA C-tile. Double-buffered LDS im2col (parity = s&1), 1 barrier/K-step.
// Gather loads are UNCONDITIONAL (clamped offset * {0,1} mask) so they batch
// and their wait lands after the WMMA instead of per-load exec-mask branches.
// ---------------------------------------------------------------------------
__global__ __launch_bounds__(128)
void conv_snn_fused(const float* __restrict__ x, const _Float16* __restrict__ wp,
                    const float* __restrict__ bias, float* __restrict__ out) {
    __shared__ __align__(16) _Float16 smA[2][16 * SMSTR];   // 2 x (16 px x 32 K)

    const int tid  = threadIdx.x;
    const int lane = tid & 31;
    const int wave = tid >> 5;                  // 0..3
    const int blk  = blockIdx.x;                // 64*4*2 = 512
    const int w0   = (blk & 3) * 16;
    const int h    = (blk >> 2) & (HW - 1);
    const int coH  = blk >> 8;                  // 0 or 1 (co half)
    const int lhi  = lane >> 4;                 // half-wave
    const int llo  = lane & 15;
    const int co   = (coH * 4 + wave) * 16 + llo;

    // ---- per-thread im2col gather geometry (loop-invariant: h, w0 fixed) ----
    // Thread handles 2 K-pairs: pp = tid + e*128 -> pixel m, even k "kp".
    int   ldsOff[2];                 // LDS half-index for the pair
    int   cioff[2];                  // 0/1: which ci of the step's channel pair
    int   off0[2],  off1[2];         // clamped offsets into x channel plane
    float msk0[2],  msk1[2];         // 0.0 / 1.0 validity masks
    #pragma unroll
    for (int e = 0; e < 2; ++e) {
        const int pp = tid + e * 128;
        const int m  = pp >> 4;
        const int kp = (pp & 15) * 2;
        ldsOff[e] = m * SMSTR + kp;
        cioff[e]  = kp >> 4;
        const int r0 = kp & 15, r1 = r0 + 1;
        off0[e] = 0; msk0[e] = 0.0f;
        off1[e] = 0; msk1[e] = 0.0f;
        if (r0 < 9) {
            int hin = h + r0 / 3 - 1, win = w0 + m + r0 % 3 - 1;
            if ((unsigned)hin < (unsigned)HW && (unsigned)win < (unsigned)HW) {
                off0[e] = hin * HW + win; msk0[e] = 1.0f;
            }
        }
        if (r1 < 9) {
            int hin = h + r1 / 3 - 1, win = w0 + m + r1 % 3 - 1;
            if ((unsigned)hin < (unsigned)HW && (unsigned)win < (unsigned)HW) {
                off1[e] = hin * HW + win; msk1[e] = 1.0f;
            }
        }
    }

    // unconditional gather of the 4 floats for flattened step q
    auto stage_load = [&](int q, float* v) {
        const int t = q >> 5, s = q & 31;
        #pragma unroll
        for (int e = 0; e < 2; ++e) {
            const size_t cbase = ((size_t)t * CIN + (2 * s + cioff[e])) * (HW * HW);
            v[2 * e + 0] = x[cbase + off0[e]] * msk0[e];
            v[2 * e + 1] = x[cbase + off1[e]] * msk1[e];
        }
    };
    // cvt + packed 32-bit LDS store into buffer q&1
    auto stage_store = [&](int q, const float* v) {
        const int buf = q & 1;
        #pragma unroll
        for (int e = 0; e < 2; ++e)
            *(h2*)(&smA[buf][ldsOff[e]]) =
                (h2){(_Float16)v[2 * e + 0], (_Float16)v[2 * e + 1]};
    };

    // ---- persistent per-lane state ----
    const float bb = bias[co];
    v8f bias8;
    #pragma unroll
    for (int j = 0; j < 8; ++j) bias8[j] = bb;
    v8f state = {};

    const int mb = lhi * 8;
    float* outBase = out + (((size_t)co * HW + h) * HW + w0 + mb);
    const _Float16* wpB = wp + (size_t)co * (CIN * KPAD) + lhi * KPAD;  // + s*32 per step

    {   // prologue: stage step 0
        float v[4];
        stage_load(0, v);
        stage_store(0, v);
    }
    __syncthreads();

    for (int t = 0; t < T_DIM; ++t) {
        v8f acc = bias8;                       // bias folded into the accumulator
        for (int s = 0; s < KSTEPS; ++s) {
            // A fragment from LDS buffer s&1 (contiguous K order, same as B):
            // lane<16: M=lane, K 0-15 ; lane>=16: M=lane-16, K 16-31
            const _Float16* ap = &smA[s & 1][llo * SMSTR + lhi * 16];
            v8h alo = *(const v8h*)(ap);
            v8h ahi = *(const v8h*)(ap + 8);
            v16h a  = __builtin_shufflevector(alo, ahi,
                                              0, 1, 2, 3, 4, 5, 6, 7,
                                              8, 9, 10, 11, 12, 13, 14, 15);

            // B fragment: L2-hot, reused across all 128 timesteps
            v16h bm = *(const v16h*)(wpB + (size_t)s * 32);

            // software pipeline: issue next step's gather loads BEFORE the WMMA,
            // convert + LDS-store them AFTER, so the wait hides under the MMA.
            const int qn = (t * KSTEPS + s) + 1;
            float nv[4];
            if (qn < QTOT) stage_load(qn, nv);

            acc = __builtin_amdgcn_wmma_f32_16x16x32_f16(false, a, false, bm,
                                                         (short)0, acc, false, false);

            if (qn < QTOT) stage_store(qn, nv);
            __syncthreads();
        }

        // ---- per-timestep epilogue: integrate + multi-spike subtract-and-fire
        v8f sp;
        #pragma unroll
        for (int j = 0; j < 8; ++j) {
            float sv = state[j] + acc[j];
            float n  = 0.0f;
            if (sv >= 8.0f) {
                n  = floorf((sv - 8.0f) * 0.125f) + 1.0f;
                sv -= n * 8.0f;
            }
            state[j] = sv;
            sp[j] = n;
        }
        *(v8f*)(outBase + (size_t)t * CHW) = sp;   // 32B contiguous per lane
    }
}

// ---------------------------------------------------------------------------
extern "C" void kernel_launch(void* const* d_in, const int* in_sizes, int n_in,
                              void* d_out, int out_size, void* d_ws, size_t ws_size,
                              hipStream_t stream) {
    const float* x  = (const float*)d_in[0];   // [128,64,64,64]
    const float* w  = (const float*)d_in[1];   // [128,64,3,3]
    const float* b  = (const float*)d_in[2];   // [128]
    float*    out   = (float*)d_out;           // [128,128,64,64]
    _Float16* wpack = (_Float16*)d_ws;         // 131072 f16 = 256 KB scratch

    pack_weights<<<(COUT * CIN * KPAD + 255) / 256, 256, 0, stream>>>(w, wpack);
    conv_snn_fused<<<HW * 4 * 2, 128, 0, stream>>>(x, wpack, b, out);
}
